// LocalRelationalLayer_1795296330401
// MI455X (gfx1250) — matrix operand order repaired and checked by reading, not compiled, not run on hardware
//
#include <hip/hip_runtime.h>

// ---------------- problem constants (from reference) ----------------
#define BB   4
#define CC   128
#define HH   56
#define WW   56
#define LL   (HH * WW)        // 3136
#define KK7  7
#define KK49 49
#define PADm 3
#define CQ   16               // C / M
#define CH8  8
#define OO   128
#define NPIX (BB * LL)        // 12544

typedef __attribute__((ext_vector_type(16))) _Float16 v16h;
typedef __attribute__((ext_vector_type(8)))  _Float16 v8h;
typedef __attribute__((ext_vector_type(8)))  float    v8f;

union V16U { v16h v; v8h h[2]; };

// ---------------------------------------------------------------
// K0: tiny prep — gpk MLP (16x49) + weight conversions to f16
// ---------------------------------------------------------------
__global__ void lrl_prep(const float* __restrict__ gp_pos,
                         const float* __restrict__ l1w, const float* __restrict__ l1b,
                         const float* __restrict__ l2w, const float* __restrict__ l2b,
                         const float* __restrict__ kmap_w, const float* __restrict__ qmap_w,
                         const float* __restrict__ final_w,
                         float* __restrict__ gpk,
                         _Float16* __restrict__ kqw16,   // [32][128] rows 0..15 kmap, 16..31 qmap
                         _Float16* __restrict__ fw16)    // [128][128]
{
    const int t = threadIdx.x;
    // gpk[cq][kk] = l2 @ relu(l1 @ pos + b1) + b2
    for (int idx = t; idx < CQ * KK49; idx += blockDim.x) {
        const int cq = idx / KK49;
        const int kk = idx % KK49;
        const float p0 = gp_pos[kk];            // channel 0
        const float p1 = gp_pos[KK49 + kk];     // channel 1
        float acc = l2b[cq];
        #pragma unroll
        for (int ch = 0; ch < CH8; ++ch) {
            float h = l1w[ch * 2 + 0] * p0 + l1w[ch * 2 + 1] * p1 + l1b[ch];
            h = fmaxf(h, 0.0f);
            acc += l2w[cq * CH8 + ch] * h;
        }
        gpk[idx] = acc;
    }
    // kmap/qmap weights -> f16, stacked [32][128]
    for (int i = t; i < CQ * CC; i += blockDim.x) {
        kqw16[i]            = (_Float16)kmap_w[i];
        kqw16[CQ * CC + i]  = (_Float16)qmap_w[i];
    }
    // final weights -> f16
    for (int i = t; i < OO * CC; i += blockDim.x) {
        fw16[i] = (_Float16)final_w[i];
    }
}

// ---------------------------------------------------------------
// K1: x (B,C,H,W) f32 -> xt[(b*L + p)*C + c] f16   (LDS tile transpose)
// grid: (L/16, C/16, B), block 256
// ---------------------------------------------------------------
__global__ void lrl_transpose(const float* __restrict__ x, _Float16* __restrict__ xt)
{
    __shared__ float tile[16][17];
    const int tx = threadIdx.x & 15;
    const int ty = threadIdx.x >> 4;
    const int p0 = blockIdx.x * 16;
    const int c0 = blockIdx.y * 16;
    const int b  = blockIdx.z;
    // coalesced read along p
    tile[ty][tx] = x[((size_t)(b * CC + c0 + ty)) * LL + p0 + tx];
    __syncthreads();
    // coalesced write along c
    xt[((size_t)(b * LL + p0 + ty)) * CC + c0 + tx] = (_Float16)tile[tx][ty];
}

// ---------------------------------------------------------------
// K2: km/qm projection via WMMA f32<=f16. One wave per 16-pixel tile.
// km/qm layout: [b][cq][l] (f32)
// grid: 98 blocks x 256 threads = 784 waves = NPIX/16 tiles
// ---------------------------------------------------------------
__global__ void lrl_proj(const _Float16* __restrict__ xt,
                         const _Float16* __restrict__ kqw16,
                         const float* __restrict__ kmap_b,
                         const float* __restrict__ qmap_b,
                         float* __restrict__ km, float* __restrict__ qm)
{
    const int wave = (blockIdx.x * blockDim.x + threadIdx.x) >> 5; // pixel tile
    const int lane = threadIdx.x & 31;
    const int col  = lane & 15;   // B-matrix column / A-matrix row
    const int hi   = lane >> 4;   // K-half selector
    const int pix  = wave * 16 + col;

    v8f acc_k = {};
    v8f acc_q = {};

    const _Float16* brow  = xt + (size_t)pix * CC + hi * 8;
    const _Float16* arowK = kqw16 + (size_t)col * CC + hi * 8;
    const _Float16* arowQ = kqw16 + (size_t)(CQ + col) * CC + hi * 8;

    #pragma unroll
    for (int k = 0; k < CC; k += 32) {
        V16U bu; bu.h[0] = *(const v8h*)(brow + k); bu.h[1] = *(const v8h*)(brow + k + 16); v16h bf = bu.v;
        V16U ak; ak.h[0] = *(const v8h*)(arowK + k); ak.h[1] = *(const v8h*)(arowK + k + 16);
        V16U aq; aq.h[0] = *(const v8h*)(arowQ + k); aq.h[1] = *(const v8h*)(arowQ + k + 16);
        acc_k = __builtin_amdgcn_wmma_f32_16x16x32_f16(false, ak.v, false, bf, (short)0, acc_k, false, false);
        acc_q = __builtin_amdgcn_wmma_f32_16x16x32_f16(false, aq.v, false, bf, (short)0, acc_q, false, false);
    }

    const int b = pix / LL;     // tiles never straddle a batch (L % 16 == 0)
    const int l = pix % LL;
    #pragma unroll
    for (int i = 0; i < 8; ++i) {
        const int m = hi * 8 + i;    // D layout: vgpr i, lanes 0-15 -> M=i ; lanes 16-31 -> M=8+i
        km[((size_t)(b * CQ + m)) * LL + l] = acc_k[i] + kmap_b[m];
        qm[((size_t)(b * CQ + m)) * LL + l] = acc_q[i] + qmap_b[m];
    }
}

// ---------------------------------------------------------------
// K3: attention (softmax over 49) + aggregation -> pre f16 [(b*L+l)*C + c]
// grid: NPIX blocks x 128 threads
// ---------------------------------------------------------------
__global__ void lrl_attn_agg(const float* __restrict__ km,
                             const float* __restrict__ qm,
                             const float* __restrict__ gpk,
                             const _Float16* __restrict__ xt,
                             _Float16* __restrict__ pre)
{
    __shared__ float ck[CQ][KK49 + 1];
    const int flat = blockIdx.x;
    const int b  = flat / LL;
    const int l  = flat % LL;
    const int ho = l / WW;
    const int wo = l % WW;
    const int t  = threadIdx.x;

    if (t < CQ) {
        const int g = t;
        const float q = qm[((size_t)(b * CQ + g)) * LL + l];
        const float* kmrow = km + ((size_t)(b * CQ + g)) * LL;
        const float* grow  = gpk + g * KK49;
        // pass 1: max
        float mx = -3.4e38f;
        {
            int kk = 0;
            for (int kr = 0; kr < KK7; ++kr) {
                const int nh = ho + kr - PADm;
                const bool rowok = (unsigned)nh < (unsigned)HH;
                for (int kc = 0; kc < KK7; ++kc, ++kk) {
                    const int nw = wo + kc - PADm;
                    const float kv = (rowok && (unsigned)nw < (unsigned)WW) ? kmrow[nh * WW + nw] : 0.0f;
                    mx = fmaxf(mx, kv * q + grow[kk]);
                }
            }
        }
        // pass 2: exp + sum
        float s = 0.0f;
        {
            int kk = 0;
            for (int kr = 0; kr < KK7; ++kr) {
                const int nh = ho + kr - PADm;
                const bool rowok = (unsigned)nh < (unsigned)HH;
                for (int kc = 0; kc < KK7; ++kc, ++kk) {
                    const int nw = wo + kc - PADm;
                    const float kv = (rowok && (unsigned)nw < (unsigned)WW) ? kmrow[nh * WW + nw] : 0.0f;
                    const float e = __expf(kv * q + grow[kk] - mx);
                    ck[g][kk] = e;
                    s += e;
                }
            }
        }
        const float inv = 1.0f / s;
        for (int kk = 0; kk < KK49; ++kk) ck[g][kk] *= inv;
    }
    __syncthreads();

    // aggregation: channel c uses group c % 16 (x reshaped (B, M, C/M, ...))
    const int c = t;
    const float* w = &ck[c & (CQ - 1)][0];
    float acc = 0.0f;
    int kk = 0;
    for (int kr = 0; kr < KK7; ++kr) {
        const int nh = ho + kr - PADm;
        const bool rowok = (unsigned)nh < (unsigned)HH;
        for (int kc = 0; kc < KK7; ++kc, ++kk) {
            const int nw = wo + kc - PADm;
            if (rowok && (unsigned)nw < (unsigned)WW) {
                acc += w[kk] * (float)xt[((size_t)(b * LL + nh * WW + nw)) * CC + c];
            }
        }
    }
    pre[(size_t)flat * CC + c] = (_Float16)acc;
}

// ---------------------------------------------------------------
// K4: final 1x1 conv via WMMA: out[b][o][l] = final_w @ pre + b
// grid: NPIX/16 blocks x 256 threads; wave w handles o-tile w (8 tiles)
// ---------------------------------------------------------------
__global__ void lrl_final(const _Float16* __restrict__ pre,
                          const _Float16* __restrict__ fw16,
                          const float* __restrict__ final_b,
                          float* __restrict__ out)
{
    const int ptile = blockIdx.x;
    const int wv    = threadIdx.x >> 5;   // 0..7 -> o tile
    const int lane  = threadIdx.x & 31;
    const int col   = lane & 15;
    const int hi    = lane >> 4;
    const int pix   = ptile * 16 + col;
    const int o0    = wv * 16;

    v8f acc = {};
    const _Float16* brow = pre  + (size_t)pix * CC + hi * 8;
    const _Float16* arow = fw16 + (size_t)(o0 + col) * CC + hi * 8;

    #pragma unroll
    for (int k = 0; k < CC; k += 32) {
        V16U bu; bu.h[0] = *(const v8h*)(brow + k); bu.h[1] = *(const v8h*)(brow + k + 16); v16h bf = bu.v;
        V16U a; a.h[0] = *(const v8h*)(arow + k); a.h[1] = *(const v8h*)(arow + k + 16);
        acc = __builtin_amdgcn_wmma_f32_16x16x32_f16(false, a.v, false, bf, (short)0, acc, false, false);
    }

    const int b = pix / LL;
    const int l = pix % LL;
    #pragma unroll
    for (int i = 0; i < 8; ++i) {
        const int o = o0 + hi * 8 + i;
        out[((size_t)(b * OO + o)) * LL + l] = acc[i] + final_b[o];
    }
}

// ---------------------------------------------------------------
// host launcher
// ---------------------------------------------------------------
extern "C" void kernel_launch(void* const* d_in, const int* in_sizes, int n_in,
                              void* d_out, int out_size, void* d_ws, size_t ws_size,
                              hipStream_t stream)
{
    const float* x       = (const float*)d_in[0];
    const float* kmap_w  = (const float*)d_in[1];
    const float* kmap_b  = (const float*)d_in[2];
    const float* qmap_w  = (const float*)d_in[3];
    const float* qmap_b  = (const float*)d_in[4];
    const float* gp_pos  = (const float*)d_in[5];
    const float* gp_l1_w = (const float*)d_in[6];
    const float* gp_l1_b = (const float*)d_in[7];
    const float* gp_l2_w = (const float*)d_in[8];
    const float* gp_l2_b = (const float*)d_in[9];
    const float* final_w = (const float*)d_in[10];
    const float* final_b = (const float*)d_in[11];
    float* out = (float*)d_out;

    // workspace carve-up (256B aligned regions)
    char* ws = (char*)d_ws;
    size_t off = 0;
    float*    gpk   = (float*)(ws + off);    off += 4096;                         // 16*49 f32
    _Float16* kqw16 = (_Float16*)(ws + off); off += 32 * CC * sizeof(_Float16);   // 8 KB
    _Float16* fw16  = (_Float16*)(ws + off); off += (size_t)OO * CC * sizeof(_Float16); // 32 KB
    float*    km    = (float*)(ws + off);    off += (size_t)BB * CQ * LL * sizeof(float); // 784 KB
    float*    qm    = (float*)(ws + off);    off += (size_t)BB * CQ * LL * sizeof(float); // 784 KB
    _Float16* xt    = (_Float16*)(ws + off); off += (size_t)NPIX * CC * sizeof(_Float16); // 3.06 MB
    _Float16* pre   = (_Float16*)(ws + off); off += (size_t)NPIX * CC * sizeof(_Float16); // 3.06 MB
    (void)ws_size; (void)in_sizes; (void)n_in; (void)out_size;

    lrl_prep<<<1, 256, 0, stream>>>(gp_pos, gp_l1_w, gp_l1_b, gp_l2_w, gp_l2_b,
                                    kmap_w, qmap_w, final_w, gpk, kqw16, fw16);

    lrl_transpose<<<dim3(LL / 16, CC / 16, BB), 256, 0, stream>>>(x, xt);

    lrl_proj<<<NPIX / 16 / 8, 256, 0, stream>>>(xt, kqw16, kmap_b, qmap_b, km, qm);

    lrl_attn_agg<<<NPIX, 128, 0, stream>>>(km, qm, gpk, xt, pre);

    lrl_final<<<NPIX / 16, 256, 0, stream>>>(pre, fw16, final_b, out);
}
